// ConditionalRoutedAttention_67310727463358
// MI455X (gfx1250) — matrix operand order repaired and checked by reading, not compiled, not run on hardware
//
#include <hip/hip_runtime.h>
#include <hip/hip_bf16.h>
#include <math.h>

#define DIM_     1024
#define HEADS_   8
#define DH_      64
#define HDH_     512
#define SEQ_     8192
#define BATCH_   2
#define WIN_     64
#define NW_      (SEQ_/WIN_)
#define NQ_      1024
#define NKV_     2048
#define NKPAD_   2176      // 17*128: 1 null + 2048 real + pad
#define NKVALID_ 2049

typedef _Float16 half_t;
typedef __attribute__((ext_vector_type(16))) _Float16 v16h;
typedef __attribute__((ext_vector_type(8)))  _Float16 v8h;
typedef __attribute__((ext_vector_type(8)))  float    v8f;

static __device__ __forceinline__ v8f zero8f() {
  v8f z;
#pragma unroll
  for (int i = 0; i < 8; ++i) z[i] = 0.0f;
  return z;
}

static __device__ __forceinline__ v8f wmma_f16(v16h a, v16h b, v8f c) {
  // D = A(16x32) * B(32x16) + C, f32 accumulate
  return __builtin_amdgcn_wmma_f32_16x16x32_f16(false, a, false, b, (short)0, c, false, false);
}

// A fragment (16x32 f16) from row-major MxK tile. ISA layout:
// lanes 0-15 row=lane: halfs0-7 = K0..7, halfs8-15 = K16..23
// lanes16-31 row=lane-16: halfs0-7 = K8..15, halfs8-15 = K24..31
static __device__ __forceinline__ v16h load_frag_a(const half_t* p, int ld) {
  const int lane = threadIdx.x & 31;
  const half_t* r = p + (lane & 15) * ld + ((lane >> 4) << 3);
  v8h lo = *(const v8h*)(r);
  v8h hi = *(const v8h*)(r + 16);
  v16h f;
#pragma unroll
  for (int i = 0; i < 8; ++i) { f[i] = lo[i]; f[i + 8] = hi[i]; }
  return f;
}

// B fragment (32x16 f16) loaded from B^T stored row-major (N x K).
// lane&15 = n; lanes0-15 hold K=0..7,16..23, lanes16-31 hold K=8..15,24..31.
static __device__ __forceinline__ v16h load_frag_b(const half_t* p, int ld) {
  const int lane = threadIdx.x & 31;
  const half_t* r = p + (lane & 15) * ld + ((lane >> 4) << 3);
  v8h lo = *(const v8h*)(r);
  v8h hi = *(const v8h*)(r + 16);
  v16h f;
#pragma unroll
  for (int i = 0; i < 8; ++i) { f[i] = lo[i]; f[i + 8] = hi[i]; }
  return f;
}

// ---- CDNA5 async global->LDS copy (ASYNCcnt-tracked, 16B per lane) ----
// VDST VGPR holds the workgroup-relative LDS byte address (== AS(3) pointer value),
// VADDR holds the 64-bit global address.  See CDNA5 ISA.
static __device__ __forceinline__ void async_copy_b128(void* lds, const void* g) {
  asm volatile("global_load_async_to_lds_b128 %0, %1, off"
               :: "v"((__attribute__((address_space(3))) void*)lds), "v"(g)
               : "memory");
}
static __device__ __forceinline__ void async_wait_all() {
  asm volatile("s_wait_asynccnt 0" ::: "memory");
}

// ---------------- transpose + convert: out[N x K] = (f16) in[K x N]^T ----------------
__global__ void k_f32_to_f16_t(const float* __restrict__ in, half_t* __restrict__ out,
                               int K, int N) {
  int i = blockIdx.x * 256 + threadIdx.x;
  if (i < K * N) {
    int n = i / K, k = i - n * K;
    out[i] = (half_t)in[(size_t)k * N + n];
  }
}

// ---------------- token stats: layernorm + rms factor + routing scores ----------------
__global__ __launch_bounds__(256) void k_token_stats(
    const float* __restrict__ x, const float* __restrict__ ln_g, const float* __restrict__ ln_b,
    const float* __restrict__ rt_q, const float* __restrict__ rt_kv,
    half_t* __restrict__ xln, float* __restrict__ sq, float* __restrict__ skv, float* __restrict__ rfac) {
  const int tok = blockIdx.x;
  const int tid = threadIdx.x;
  const float* xr = x + (size_t)tok * DIM_;
  float4 xv = *(const float4*)(xr + tid * 4);
  float4 rq = *(const float4*)(rt_q + tid * 4);
  float4 rk = *(const float4*)(rt_kv + tid * 4);
  float s  = xv.x + xv.y + xv.z + xv.w;
  float ss = xv.x * xv.x + xv.y * xv.y + xv.z * xv.z + xv.w * xv.w;
  float dq = xv.x * rq.x + xv.y * rq.y + xv.z * rq.z + xv.w * rq.w;
  float dk = xv.x * rk.x + xv.y * rk.y + xv.z * rk.z + xv.w * rk.w;
  __shared__ float red[4 * 256];
  __shared__ float mv[2];
  red[tid] = s; red[256 + tid] = ss; red[512 + tid] = dq; red[768 + tid] = dk;
  __syncthreads();
  for (int st = 128; st > 0; st >>= 1) {
    if (tid < st) {
      red[tid]       += red[tid + st];
      red[256 + tid] += red[256 + tid + st];
      red[512 + tid] += red[512 + tid + st];
      red[768 + tid] += red[768 + tid + st];
    }
    __syncthreads();
  }
  if (tid == 0) {
    float sum = red[0], sumsq = red[256];
    float mean = sum * (1.0f / DIM_);
    float var  = sumsq * (1.0f / DIM_) - mean * mean;
    mv[0] = mean;
    mv[1] = rsqrtf(var + 1e-5f);
    float nrm = fmaxf(sqrtf(sumsq), 1e-12f);
    rfac[tok] = 32.0f / nrm;   // sqrt(DIM)=32
    sq[tok]  = red[512];
    skv[tok] = red[768];
  }
  __syncthreads();
  float mean = mv[0], rstd = mv[1];
  half_t* o = xln + (size_t)tok * DIM_;
  int d = tid * 4;
  o[d + 0] = (half_t)((xv.x - mean) * rstd * ln_g[d + 0] + ln_b[d + 0]);
  o[d + 1] = (half_t)((xv.y - mean) * rstd * ln_g[d + 1] + ln_b[d + 1]);
  o[d + 2] = (half_t)((xv.z - mean) * rstd * ln_g[d + 2] + ln_b[d + 2]);
  o[d + 3] = (half_t)((xv.w - mean) * rstd * ln_g[d + 3] + ln_b[d + 3]);
}

// ---------------- generic f16 WMMA GEMM: C = A(MxK) @ B, with B given as B^T (NxK) ---------
// MODE 0: store f16 to Ch. MODE 1: store f32 (+ optional bias[col]) to Cf.
// A/B tiles staged to LDS with CDNA5 async global->LDS (ASYNCcnt).
template <int MODE>
__global__ __launch_bounds__(256) void k_gemm(
    const half_t* __restrict__ A, const half_t* __restrict__ BT,
    half_t* __restrict__ Ch, float* __restrict__ Cf, const float* __restrict__ bias,
    int M, int N, int K) {
  const int LDA = 40;
  __shared__ half_t As[128 * 40];
  __shared__ half_t Bs[128 * 40];   // B^T tile (n-major)
  const int tid = threadIdx.x;
  const int bm = blockIdx.y * 128, bn = blockIdx.x * 128;
  const int wid = tid >> 5;
  const int wm = (wid >> 2) * 64, wn = (wid & 3) * 32;
  const int lane = tid & 31, cl = lane & 15, rh = ((lane >> 4) << 3);
  v8f acc[4][2];
#pragma unroll
  for (int i = 0; i < 4; ++i)
#pragma unroll
    for (int j = 0; j < 2; ++j) acc[i][j] = zero8f();
  for (int k0 = 0; k0 < K; k0 += 32) {
    __syncthreads();  // all waves done reading As/Bs from previous step
#pragma unroll
    for (int i = 0; i < 2; ++i) {
      int sA = tid + i * 256;            // 512 x 16B segments: row 0..127, 4 segs/row
      int row = sA >> 2, c = (sA & 3) * 8;
      async_copy_b128(As + row * LDA + c, A  + (size_t)(bm + row) * K + k0 + c);
      async_copy_b128(Bs + row * LDA + c, BT + (size_t)(bn + row) * K + k0 + c);
    }
    __builtin_prefetch(A  + (size_t)(bm + (tid >> 2)) * K + k0 + 32, 0, 1);  // global_prefetch_b8
    __builtin_prefetch(BT + (size_t)(bn + (tid >> 2)) * K + k0 + 32, 0, 1);
    async_wait_all();
    __syncthreads();
    v16h af[4], bf[2];
#pragma unroll
    for (int mt = 0; mt < 4; ++mt) af[mt] = load_frag_a(As + (wm + mt * 16) * LDA, LDA);
#pragma unroll
    for (int nt = 0; nt < 2; ++nt) bf[nt] = load_frag_b(Bs + (wn + nt * 16) * LDA, LDA);
#pragma unroll
    for (int mt = 0; mt < 4; ++mt)
#pragma unroll
      for (int nt = 0; nt < 2; ++nt) acc[mt][nt] = wmma_f16(af[mt], bf[nt], acc[mt][nt]);
  }
#pragma unroll
  for (int mt = 0; mt < 4; ++mt)
#pragma unroll
    for (int nt = 0; nt < 2; ++nt)
#pragma unroll
      for (int r = 0; r < 8; ++r) {
        int row = bm + wm + mt * 16 + rh + r;
        int col = bn + wn + nt * 16 + cl;
        float v = acc[mt][nt][r];
        if (MODE == 0) Ch[(size_t)row * N + col] = (half_t)v;
        else           Cf[(size_t)row * N + col] = v + (bias ? bias[col] : 0.0f);
      }
}

// ---------------- light windowed attention ----------------
__global__ __launch_bounds__(128) void k_light_attn(const half_t* __restrict__ qkv,
                                                    half_t* __restrict__ lightO) {
  const int w = blockIdx.x, h = blockIdx.y, b = blockIdx.z;
  const int tid = threadIdx.x, wid = tid >> 5, lane = tid & 31, cl = lane & 15, rh = ((lane >> 4) << 3);
  const int LQ = 72, LK = 72, LV = 200;
  __shared__ half_t qs[64 * 72];
  __shared__ half_t ks[192 * 72];
  __shared__ half_t vts[64 * 200];
  half_t* ps = ks;  // P (64x200 <= 192x72) overlays K after barrier
  const size_t rowbase = (size_t)b * SEQ_;
  for (int idx = tid; idx < 64 * 8; idx += 128) {
    int r = idx >> 3, c0 = (idx & 7) * 8;
    *(v8h*)(qs + r * LQ + c0) =
        *(const v8h*)(qkv + (rowbase + (size_t)(w * WIN_ + r)) * 1536 + h * 64 + c0);
  }
  const bool v0 = (w > 0), v2 = (w < NW_ - 1);
  for (int idx = tid; idx < 192 * 8; idx += 128) {
    int n = idx >> 3, c0 = (idx & 7) * 8;
    int seg = n >> 6;
    bool valid = (seg == 1) || (seg == 0 && v0) || (seg == 2 && v2);
    v8h kv, vv;
    if (valid) {
      size_t row = rowbase + (size_t)((w - 1) * WIN_ + n);
      kv = *(const v8h*)(qkv + row * 1536 + 512  + h * 64 + c0);
      vv = *(const v8h*)(qkv + row * 1536 + 1024 + h * 64 + c0);
    } else {
#pragma unroll
      for (int i = 0; i < 8; ++i) { kv[i] = (half_t)0.0f; vv[i] = (half_t)0.0f; }
    }
    *(v8h*)(ks + n * LK + c0) = kv;
#pragma unroll
    for (int i = 0; i < 8; ++i) vts[(c0 + i) * LV + n] = vv[i];
  }
  __syncthreads();
  v8f sim[12];
#pragma unroll
  for (int nt = 0; nt < 12; ++nt) sim[nt] = zero8f();
  v16h a0 = load_frag_a(qs + wid * 16 * LQ, LQ);
  v16h a1 = load_frag_a(qs + wid * 16 * LQ + 32, LQ);
#pragma unroll
  for (int nt = 0; nt < 12; ++nt) {
    v16h b0 = load_frag_b(ks + nt * 16 * LK, LK);
    v16h b1 = load_frag_b(ks + nt * 16 * LK + 32, LK);
    sim[nt] = wmma_f16(a0, b0, sim[nt]);
    sim[nt] = wmma_f16(a1, b1, sim[nt]);
  }
  float mrow[8], lrow[8];
#pragma unroll
  for (int r = 0; r < 8; ++r) mrow[r] = -3.0e38f;
#pragma unroll
  for (int nt = 0; nt < 12; ++nt) {
    int col = nt * 16 + cl, seg = col >> 6;
    bool valid = (seg == 1) || (seg == 0 && v0) || (seg == 2 && v2);
#pragma unroll
    for (int r = 0; r < 8; ++r) {
      float sv = sim[nt][r] * 0.125f;          // DH^-0.5
      sv = valid ? sv : -3.0e38f;
      sim[nt][r] = sv;
      mrow[r] = fmaxf(mrow[r], sv);
    }
  }
#pragma unroll
  for (int off = 1; off < 16; off <<= 1)
#pragma unroll
    for (int r = 0; r < 8; ++r) mrow[r] = fmaxf(mrow[r], __shfl_xor(mrow[r], off, 32));
#pragma unroll
  for (int r = 0; r < 8; ++r) lrow[r] = 0.0f;
#pragma unroll
  for (int nt = 0; nt < 12; ++nt)
#pragma unroll
    for (int r = 0; r < 8; ++r) {
      float p = __expf(sim[nt][r] - mrow[r]);
      sim[nt][r] = p;
      lrow[r] += p;
    }
#pragma unroll
  for (int off = 1; off < 16; off <<= 1)
#pragma unroll
    for (int r = 0; r < 8; ++r) lrow[r] += __shfl_xor(lrow[r], off, 32);
  __syncthreads();  // all waves done reading ks before P overlays it
#pragma unroll
  for (int nt = 0; nt < 12; ++nt)
#pragma unroll
    for (int r = 0; r < 8; ++r)
      ps[(wid * 16 + rh + r) * LV + nt * 16 + cl] = (half_t)sim[nt][r];
  v8f oacc[4];
#pragma unroll
  for (int nt = 0; nt < 4; ++nt) oacc[nt] = zero8f();
#pragma unroll
  for (int kc = 0; kc < 6; ++kc) {
    v16h ap = load_frag_a(ps + wid * 16 * LV + kc * 32, LV);
#pragma unroll
    for (int nt = 0; nt < 4; ++nt) {
      v16h bv = load_frag_b(vts + nt * 16 * LV + kc * 32, LV);
      oacc[nt] = wmma_f16(ap, bv, oacc[nt]);
    }
  }
#pragma unroll
  for (int nt = 0; nt < 4; ++nt)
#pragma unroll
    for (int r = 0; r < 8; ++r)
      lightO[(rowbase + (size_t)(w * WIN_ + wid * 16 + rh + r)) * HDH_ + h * 64 + nt * 16 + cl]
          = (half_t)(oacc[nt][r] / lrow[r]);
}

// ---------------- coordinate descent (EPS=1) ----------------
__global__ __launch_bounds__(1024) void k_coor(const float* __restrict__ sqp, const float* __restrict__ skvp,
                                               float* __restrict__ outq, float* __restrict__ outkv,
                                               float logk_q, float logk_kv) {
  const int b = blockIdx.x, type = blockIdx.y;
  const float* s = (type == 0 ? sqp : skvp) + (size_t)b * SEQ_;
  float* out = (type == 0 ? outq : outkv) + (size_t)b * SEQ_;
  const float logk = (type == 0) ? logk_q : logk_kv;
  __shared__ float red[1024];
  const int tid = threadIdx.x;
  float sv[8], bv[8];
#pragma unroll
  for (int i = 0; i < 8; ++i) { sv[i] = s[tid + i * 1024]; bv[i] = -sv[i]; }
  float a = 0.0f;
  for (int it = 0; it < 50; ++it) {
    float lm = -3.0e38f;
#pragma unroll
    for (int i = 0; i < 8; ++i) lm = fmaxf(lm, sv[i] + bv[i]);
    red[tid] = lm;
    __syncthreads();
    for (int st = 512; st > 0; st >>= 1) { if (tid < st) red[tid] = fmaxf(red[tid], red[tid + st]); __syncthreads(); }
    float m = red[0];
    __syncthreads();
    float ls = 0.0f;
#pragma unroll
    for (int i = 0; i < 8; ++i) ls += __expf(sv[i] + bv[i] - m);
    red[tid] = ls;
    __syncthreads();
    for (int st = 512; st > 0; st >>= 1) { if (tid < st) red[tid] += red[tid + st]; __syncthreads(); }
    float ssum = red[0];
    __syncthreads();
    a = logk - (m + __logf(ssum));
#pragma unroll
    for (int i = 0; i < 8; ++i) bv[i] = -fmaxf(sv[i] + a, 0.0f);
  }
#pragma unroll
  for (int i = 0; i < 8; ++i) out[tid + i * 1024] = __expf(sv[i] + a + bv[i]);
}

// ---------------- bitonic top-k (descending score, ascending index on ties) ----------------
__global__ __launch_bounds__(1024) void k_topk(const float* __restrict__ scq, const float* __restrict__ sckv,
                                               int* __restrict__ iq, int* __restrict__ ikv) {
  const int b = blockIdx.x, type = blockIdx.y;
  const float* sc = (type ? sckv : scq) + (size_t)b * SEQ_;
  const int K = type ? NKV_ : NQ_;
  int* out = type ? (ikv + b * NKV_) : (iq + b * NQ_);
  __shared__ unsigned long long keys[SEQ_];
  const int tid = threadIdx.x;
  for (int i = tid; i < SEQ_; i += 1024) {
    unsigned fb = __float_as_uint(sc[i]);  // scores > 0 => uint order == float order
    keys[i] = ((unsigned long long)fb << 32) | (unsigned)(SEQ_ - 1 - i);
  }
  __syncthreads();
  for (unsigned k = 2; k <= (unsigned)SEQ_; k <<= 1)
    for (unsigned j = k >> 1; j > 0; j >>= 1) {
      for (int e = 0; e < SEQ_ / 1024; ++e) {
        unsigned i = (unsigned)tid + (unsigned)e * 1024u;
        unsigned ixj = i ^ j;
        if (ixj > i) {
          unsigned long long A = keys[i], B = keys[ixj];
          bool up = ((i & k) == 0);
          if (up ? (A < B) : (A > B)) { keys[i] = B; keys[ixj] = A; }
        }
      }
      __syncthreads();
    }
  for (int jx = tid; jx < K; jx += 1024)
    out[jx] = SEQ_ - 1 - (int)(keys[jx] & 0xFFFFFFFFull);
}

// ---------------- gather + rmsnorm -> f16 ----------------
__global__ void k_gather(const float* __restrict__ x, const int* __restrict__ idx,
                         const float* __restrict__ rfac, const float* __restrict__ rms_g,
                         half_t* __restrict__ out, int NTOK) {
  const int j = blockIdx.x, b = blockIdx.y;
  const int t = idx[b * NTOK + j];
  const float* xr = x + ((size_t)b * SEQ_ + t) * DIM_;
  const float f = rfac[b * SEQ_ + t];
  half_t* o = out + ((size_t)b * NTOK + j) * DIM_;
  for (int d = threadIdx.x; d < DIM_; d += 256) o[d] = (half_t)(xr[d] * f * rms_g[d]);
}

// ---------------- rearrange kv: prepend null kv, pad, build K rows + V^T ----------------
__global__ __launch_bounds__(64) void k_rearrange_kv(const half_t* __restrict__ kvraw,
                                                     const float* __restrict__ null_kv,
                                                     half_t* __restrict__ kH, half_t* __restrict__ vT) {
  const int n = blockIdx.x, h = blockIdx.y, b = blockIdx.z, d = threadIdx.x;
  half_t kv, vv;
  if (n == 0) {
    kv = (half_t)null_kv[(0 * HEADS_ + h) * DH_ + d];
    vv = (half_t)null_kv[(1 * HEADS_ + h) * DH_ + d];
  } else if (n <= NKV_) {
    const half_t* src = kvraw + ((size_t)(b * NKV_ + n - 1)) * (HEADS_ * 2 * DH_) + h * 128;
    kv = src[d];
    vv = src[64 + d];
  } else {
    kv = (half_t)0.0f;
    vv = (half_t)0.0f;
  }
  kH[((size_t)(b * HEADS_ + h) * NKPAD_ + n) * DH_ + d] = kv;
  vT[((size_t)(b * HEADS_ + h) * DH_ + d) * NKPAD_ + n] = vv;
}

// ---------------- heavy routed attention (flash-style online softmax) ----------------
__global__ __launch_bounds__(128) void k_heavy_attn(const half_t* __restrict__ qraw,
                                                    const half_t* __restrict__ kH,
                                                    const half_t* __restrict__ vT,
                                                    half_t* __restrict__ heavyO) {
  const int qt = blockIdx.x, h = blockIdx.y, b = blockIdx.z;
  const int tid = threadIdx.x, wid = tid >> 5, lane = tid & 31, cl = lane & 15, rh = ((lane >> 4) << 3);
  const int LQ = 72, LK = 72, LV = 136;
  __shared__ half_t qs[64 * 72];
  __shared__ half_t ks[128 * 72];
  __shared__ half_t vts[64 * 136];
  __shared__ half_t ps[64 * 136];
  // async-stage Q tile (64x64): 512 x 16B segments, 4 per thread
#pragma unroll
  for (int i = 0; i < 4; ++i) {
    int s = tid + i * 128;
    int r = s >> 3, c0 = (s & 7) * 8;
    async_copy_b128(qs + r * LQ + c0,
                    qraw + ((size_t)(b * NQ_) + qt * 64 + r) * HDH_ + h * 64 + c0);
  }
  const half_t* kb_ = kH + (size_t)(b * HEADS_ + h) * NKPAD_ * DH_;
  const half_t* vb_ = vT + (size_t)(b * HEADS_ + h) * DH_ * NKPAD_;
  float m[8], l[8];
  v8f oacc[4];
#pragma unroll
  for (int r = 0; r < 8; ++r) { m[r] = -3.0e38f; l[r] = 0.0f; }
#pragma unroll
  for (int nt = 0; nt < 4; ++nt) oacc[nt] = zero8f();
  for (int kb = 0; kb < NKPAD_ / 128; ++kb) {
    __syncthreads();  // all waves done with ks/vts of previous chunk
#pragma unroll
    for (int i = 0; i < 8; ++i) {   // K chunk 128x64: 1024 segs
      int s = tid + i * 128;
      int r = s >> 3, c0 = (s & 7) * 8;
      async_copy_b128(ks + r * LK + c0, kb_ + (size_t)(kb * 128 + r) * DH_ + c0);
    }
#pragma unroll
    for (int i = 0; i < 8; ++i) {   // V^T chunk 64x128: 1024 segs
      int s = tid + i * 128;
      int d = s >> 4, c0 = (s & 15) * 8;
      async_copy_b128(vts + d * LV + c0, vb_ + (size_t)d * NKPAD_ + kb * 128 + c0);
    }
    async_wait_all();
    __syncthreads();
    v8f sim[8];
#pragma unroll
    for (int nt = 0; nt < 8; ++nt) sim[nt] = zero8f();
    v16h a0 = load_frag_a(qs + wid * 16 * LQ, LQ);
    v16h a1 = load_frag_a(qs + wid * 16 * LQ + 32, LQ);
#pragma unroll
    for (int nt = 0; nt < 8; ++nt) {
      v16h b0 = load_frag_b(ks + nt * 16 * LK, LK);
      v16h b1 = load_frag_b(ks + nt * 16 * LK + 32, LK);
      sim[nt] = wmma_f16(a0, b0, sim[nt]);
      sim[nt] = wmma_f16(a1, b1, sim[nt]);
    }
    float cm[8];
#pragma unroll
    for (int r = 0; r < 8; ++r) cm[r] = -3.0e38f;
#pragma unroll
    for (int nt = 0; nt < 8; ++nt) {
      int col = kb * 128 + nt * 16 + cl;
      bool valid = (col < NKVALID_);
#pragma unroll
      for (int r = 0; r < 8; ++r) {
        float sv = sim[nt][r] * 0.125f;
        sv = valid ? sv : -3.0e38f;
        sim[nt][r] = sv;
        cm[r] = fmaxf(cm[r], sv);
      }
    }
#pragma unroll
    for (int off = 1; off < 16; off <<= 1)
#pragma unroll
      for (int r = 0; r < 8; ++r) cm[r] = fmaxf(cm[r], __shfl_xor(cm[r], off, 32));
    float fsc[8], rs[8];
#pragma unroll
    for (int r = 0; r < 8; ++r) {
      float nm = fmaxf(m[r], cm[r]);
      fsc[r] = __expf(m[r] - nm);
      m[r] = nm;
      rs[r] = 0.0f;
    }
#pragma unroll
    for (int nt = 0; nt < 8; ++nt)
#pragma unroll
      for (int r = 0; r < 8; ++r) {
        float p = __expf(sim[nt][r] - m[r]);
        sim[nt][r] = p;
        rs[r] += p;
      }
#pragma unroll
    for (int off = 1; off < 16; off <<= 1)
#pragma unroll
      for (int r = 0; r < 8; ++r) rs[r] += __shfl_xor(rs[r], off, 32);
#pragma unroll
    for (int r = 0; r < 8; ++r) l[r] = l[r] * fsc[r] + rs[r];
#pragma unroll
    for (int nt = 0; nt < 4; ++nt)
#pragma unroll
      for (int r = 0; r < 8; ++r) oacc[nt][r] *= fsc[r];
#pragma unroll
    for (int nt = 0; nt < 8; ++nt)
#pragma unroll
      for (int r = 0; r < 8; ++r)
        ps[(wid * 16 + rh + r) * LV + nt * 16 + cl] = (half_t)sim[nt][r];
#pragma unroll
    for (int kc = 0; kc < 4; ++kc) {
      v16h ap = load_frag_a(ps + wid * 16 * LV + kc * 32, LV);
#pragma unroll
      for (int nt = 0; nt < 4; ++nt) {
        v16h bv = load_frag_b(vts + nt * 16 * LV + kc * 32, LV);
        oacc[nt] = wmma_f16(ap, bv, oacc[nt]);
      }
    }
  }
#pragma unroll
  for (int nt = 0; nt < 4; ++nt)
#pragma unroll
    for (int r = 0; r < 8; ++r)
      heavyO[((size_t)(b * NQ_) + qt * 64 + wid * 16 + rh + r) * HDH_ + h * 64 + nt * 16 + cl]
          = (half_t)(oacc[nt][r] / l[r]);
}

// ---------------- scatter routed output ----------------
__global__ void k_scatter(const float* __restrict__ routed, const int* __restrict__ iq,
                          const float* __restrict__ null_q, float* __restrict__ out) {
  const int j = blockIdx.x, b = blockIdx.y;
  const int t = iq[b * NQ_ + j];
  float* o = out + ((size_t)b * SEQ_ + t) * DIM_;
  const float* r = routed + ((size_t)b * NQ_ + j) * DIM_;
  for (int d = threadIdx.x; d < DIM_; d += 256) o[d] += r[d] - null_q[d];
}

extern "C" void kernel_launch(void* const* d_in, const int* in_sizes, int n_in,
                              void* d_out, int out_size, void* d_ws, size_t ws_size,
                              hipStream_t stream) {
  (void)in_sizes; (void)n_in; (void)out_size; (void)ws_size;
  const float* x       = (const float*)d_in[0];
  const float* ln_g    = (const float*)d_in[1];
  const float* ln_b    = (const float*)d_in[2];
  const float* w_qkv_l = (const float*)d_in[3];
  const float* w_out_l = (const float*)d_in[4];
  const float* null_q  = (const float*)d_in[5];
  const float* rt_q    = (const float*)d_in[6];
  const float* rt_kv   = (const float*)d_in[7];
  const float* rms_g   = (const float*)d_in[8];
  const float* w_q_h   = (const float*)d_in[9];
  const float* w_kv_h  = (const float*)d_in[10];
  const float* null_kv = (const float*)d_in[11];
  const float* w_out_h = (const float*)d_in[12];
  float* out = (float*)d_out;

  char* wsb = (char*)d_ws;
  size_t off = 0;
  auto alloc = [&](size_t bytes) -> void* {
    void* p = wsb + off;
    off += (bytes + 255) & ~(size_t)255;
    return p;
  };
  const int TOK = BATCH_ * SEQ_;  // 16384
  // transposed f16 weights (N x K, row-major)
  half_t* wqkvT  = (half_t*)alloc((size_t)3 * HDH_ * DIM_ * 2);
  half_t* woutlT = (half_t*)alloc((size_t)DIM_ * HDH_ * 2);
  half_t* wqT    = (half_t*)alloc((size_t)HDH_ * DIM_ * 2);
  half_t* wkvT   = (half_t*)alloc((size_t)DIM_ * DIM_ * 2);
  half_t* wouthT = (half_t*)alloc((size_t)DIM_ * HDH_ * 2);
  half_t* xln_h  = (half_t*)alloc((size_t)TOK * DIM_ * 2);
  half_t* qkv_h  = (half_t*)alloc((size_t)TOK * 3 * HDH_ * 2);
  half_t* lightO = (half_t*)alloc((size_t)TOK * HDH_ * 2);
  float* sqb   = (float*)alloc((size_t)TOK * 4);
  float* skvb  = (float*)alloc((size_t)TOK * 4);
  float* scq   = (float*)alloc((size_t)TOK * 4);
  float* sckv  = (float*)alloc((size_t)TOK * 4);
  float* rfac  = (float*)alloc((size_t)TOK * 4);
  int* iq  = (int*)alloc((size_t)BATCH_ * NQ_ * 4);
  int* ikv = (int*)alloc((size_t)BATCH_ * NKV_ * 4);
  half_t* xq_h    = (half_t*)alloc((size_t)BATCH_ * NQ_ * DIM_ * 2);
  half_t* xkv_h   = (half_t*)alloc((size_t)BATCH_ * NKV_ * DIM_ * 2);
  half_t* qraw    = (half_t*)alloc((size_t)BATCH_ * NQ_ * HDH_ * 2);
  half_t* kvraw   = (half_t*)alloc((size_t)BATCH_ * NKV_ * DIM_ * 2);
  half_t* kHb     = (half_t*)alloc((size_t)BATCH_ * HEADS_ * NKPAD_ * DH_ * 2);
  half_t* vTb     = (half_t*)alloc((size_t)BATCH_ * HEADS_ * DH_ * NKPAD_ * 2);
  half_t* heavyO  = (half_t*)alloc((size_t)BATCH_ * NQ_ * HDH_ * 2);
  float* routedF  = (float*)alloc((size_t)BATCH_ * NQ_ * DIM_ * 4);

  auto cvtT = [&](const float* src, half_t* dst, int K, int N) {
    k_f32_to_f16_t<<<(K * N + 255) / 256, 256, 0, stream>>>(src, dst, K, N);
  };
  cvtT(w_qkv_l, wqkvT, DIM_, 3 * HDH_);
  cvtT(w_out_l, woutlT, HDH_, DIM_);
  cvtT(w_q_h,   wqT,   DIM_, HDH_);
  cvtT(w_kv_h,  wkvT,  DIM_, DIM_);
  cvtT(w_out_h, wouthT, HDH_, DIM_);

  k_token_stats<<<TOK, 256, 0, stream>>>(x, ln_g, ln_b, rt_q, rt_kv, xln_h, sqb, skvb, rfac);

  // light branch
  k_gemm<0><<<dim3((3 * HDH_) / 128, TOK / 128), 256, 0, stream>>>(
      xln_h, wqkvT, qkv_h, nullptr, nullptr, TOK, 3 * HDH_, DIM_);
  k_light_attn<<<dim3(NW_, HEADS_, BATCH_), 128, 0, stream>>>(qkv_h, lightO);
  k_gemm<1><<<dim3(DIM_ / 128, TOK / 128), 256, 0, stream>>>(
      lightO, woutlT, nullptr, out, null_q, TOK, DIM_, HDH_);

  // routing
  k_coor<<<dim3(BATCH_, 2), 1024, 0, stream>>>(sqb, skvb, scq, sckv,
                                               logf(1152.0f), logf(2304.0f));
  k_topk<<<dim3(BATCH_, 2), 1024, 0, stream>>>(scq, sckv, iq, ikv);

  // heavy branch
  k_gather<<<dim3(NQ_, BATCH_), 256, 0, stream>>>(x, iq, rfac, rms_g, xq_h, NQ_);
  k_gather<<<dim3(NKV_, BATCH_), 256, 0, stream>>>(x, ikv, rfac, rms_g, xkv_h, NKV_);
  k_gemm<0><<<dim3(HDH_ / 128, (BATCH_ * NQ_) / 128), 256, 0, stream>>>(
      xq_h, wqT, qraw, nullptr, nullptr, BATCH_ * NQ_, HDH_, DIM_);
  k_gemm<0><<<dim3(DIM_ / 128, (BATCH_ * NKV_) / 128), 256, 0, stream>>>(
      xkv_h, wkvT, kvraw, nullptr, nullptr, BATCH_ * NKV_, DIM_, DIM_);
  k_rearrange_kv<<<dim3(NKPAD_, HEADS_, BATCH_), 64, 0, stream>>>(kvraw, null_kv, kHb, vTb);
  k_heavy_attn<<<dim3(NQ_ / 64, HEADS_, BATCH_), 128, 0, stream>>>(qraw, kHb, vTb, heavyO);
  k_gemm<1><<<dim3(DIM_ / 128, (BATCH_ * NQ_) / 128), 256, 0, stream>>>(
      heavyO, wouthT, nullptr, routedF, nullptr, BATCH_ * NQ_, DIM_, HDH_);
  k_scatter<<<dim3(NQ_, BATCH_), 256, 0, stream>>>(routedF, iq, null_q, out);
}
